// OutputPPBlock_18322330485446
// MI455X (gfx1250) — compile-verified, matrix operand order checked by reading, not run on hardware
//
#include <hip/hip_runtime.h>
#include <hip/hip_bf16.h>
#include <stdint.h>

// ---------------------------------------------------------------------------
// Problem constants (from reference)
#define NUM_RADIAL 6
#define HIDDEN     128
#define OUT_EMB    256
#define OUT_CH     128
#define NUM_LAYERS 3

typedef __attribute__((ext_vector_type(8)))  _Float16 v8h;
typedef __attribute__((ext_vector_type(16))) _Float16 v16h;
typedef __attribute__((ext_vector_type(8)))  float    v8f;

// ---------------------------------------------------------------------------
// Kernel 0: prep — zero agg accumulator, convert weights fp32 -> f16
__global__ __launch_bounds__(256) void prep_kernel(
    const float* __restrict__ Wup, const float* __restrict__ Ws,
    const float* __restrict__ Wout,
    _Float16* __restrict__ WupH, _Float16* __restrict__ WsH,
    _Float16* __restrict__ WoutH,
    float* __restrict__ agg, int aggN)
{
  int tid    = blockIdx.x * blockDim.x + threadIdx.x;
  int stride = gridDim.x * blockDim.x;
  for (int t = tid; t < OUT_EMB * HIDDEN; t += stride) WupH[t]  = (_Float16)Wup[t];
  for (int t = tid; t < NUM_LAYERS * OUT_EMB * OUT_EMB; t += stride) WsH[t] = (_Float16)Ws[t];
  for (int t = tid; t < OUT_CH * OUT_EMB; t += stride)  WoutH[t] = (_Float16)Wout[t];
  for (int t = tid; t < aggN; t += stride)              agg[t]   = 0.0f;
}

// ---------------------------------------------------------------------------
// Kernel 1: edge stage.  g[e,h] = (sum_r rbf[e,r]*W_rbf[h,r]) * x[e,h]
// scatter-added into agg[i[e], h].  One thread per (edge, 4 channels).
__global__ __launch_bounds__(256) void edge_scatter_kernel(
    const float* __restrict__ x, const float* __restrict__ rbf,
    const long long* __restrict__ idx, const float* __restrict__ Wr,
    float* __restrict__ agg, int E)
{
  __shared__ float sW[HIDDEN * NUM_RADIAL];          // 3 KB, [h][r]
  for (int t = threadIdx.x; t < HIDDEN * NUM_RADIAL; t += 256) sW[t] = Wr[t];
  __syncthreads();

  long long gid = (long long)blockIdx.x * 256 + threadIdx.x;
  long long e   = gid >> 5;                          // 32 threads per edge
  if (e >= E) return;
  int hq = (int)(gid & 31) * 4;                      // 4 consecutive channels

  const float* rb = rbf + e * NUM_RADIAL;
  float r0 = rb[0], r1 = rb[1], r2 = rb[2], r3 = rb[3], r4 = rb[4], r5 = rb[5];
  long long node = idx[e];

  float4 xv = *(const float4*)(x + e * HIDDEN + hq); // global_load_b128
  float xs[4] = {xv.x, xv.y, xv.z, xv.w};
  float* dst = agg + node * HIDDEN + hq;
  const float* w = sW + hq * NUM_RADIAL;
#pragma unroll
  for (int c = 0; c < 4; ++c) {
    const float* wc = w + c * NUM_RADIAL;
    float s = r0*wc[0] + r1*wc[1] + r2*wc[2] + r3*wc[3] + r4*wc[4] + r5*wc[5];
    atomicAdd(dst + c, s * xs[c]);                   // L2-resident accumulator
  }
}

// ---------------------------------------------------------------------------
// WMMA fragment helpers (16-bit A 16x32 and B 32x16 layouts, ISA 7.12.2):
// lanes 0-15 cover K {k0..k0+7, k0+16..k0+23}; lanes 16-31 cover K {+8, +24}.
__device__ __forceinline__ v16h frag_from_lds(const _Float16* base, int lane, int k0) {
  int row  = lane & 15;
  int koff = (lane & 16) ? 8 : 0;
  const _Float16* p = base + row * OUT_EMB + k0 + koff;
  v8h lo = *(const v8h*)p;          // ds_load_b128
  v8h hi = *(const v8h*)(p + 16);   // ds_load_b128
  v16h r;
#pragma unroll
  for (int t = 0; t < 8; ++t) { r[t] = lo[t]; r[t + 8] = hi[t]; }
  return r;
}

__device__ __forceinline__ v16h frag_from_weights(const _Float16* __restrict__ W,
                                                  int K, int o0, int lane, int k0) {
  // B[k, n] = W[o0+n, k]  (torch weight [out,in], applied as x @ W^T)
  int n    = lane & 15;
  int koff = (lane & 16) ? 8 : 0;
  const _Float16* p = W + (long long)(o0 + n) * K + k0 + koff;
  v8h lo = *(const v8h*)p;          // global_load_b128 (L2-hot weights)
  v8h hi = *(const v8h*)(p + 16);
  v16h r;
#pragma unroll
  for (int t = 0; t < 8; ++t) { r[t] = lo[t]; r[t + 8] = hi[t]; }
  return r;
}

// One layer over a 32-row node tile: two 16-row M-subtiles share each B
// fragment, so every weight fetch feeds two v_wmma ops.
template<int K, int NOUT, bool HAS_BIAS, bool ACT, bool TO_LDS>
__device__ __forceinline__ void mlp_layer(const _Float16* inbuf, _Float16* outbuf,
                                          const _Float16* __restrict__ W,
                                          const float* __restrict__ bias,
                                          float* __restrict__ gout,
                                          int node0, int N, int lane)
{
  int col   = lane & 15;
  int rbase = (lane & 16) ? 8 : 0;   // D layout: lanes>=16 hold rows M=8..15
#pragma unroll 1
  for (int o0 = 0; o0 < NOUT; o0 += 16) {
    v8f acc0 = {};
    v8f acc1 = {};
#pragma unroll
    for (int k0 = 0; k0 < K; k0 += 32) {
      v16h b  = frag_from_weights(W, K, o0, lane, k0);
      v16h a0 = frag_from_lds(inbuf, lane, k0);
      v16h a1 = frag_from_lds(inbuf + 16 * OUT_EMB, lane, k0);
      acc0 = __builtin_amdgcn_wmma_f32_16x16x32_f16(false, a0, false, b,
                                                    (short)0, acc0, false, false);
      acc1 = __builtin_amdgcn_wmma_f32_16x16x32_f16(false, a1, false, b,
                                                    (short)0, acc1, false, false);
    }
    float bv = HAS_BIAS ? bias[o0 + col] : 0.0f;
#pragma unroll
    for (int m = 0; m < 2; ++m) {
#pragma unroll
      for (int j = 0; j < 8; ++j) {
        float v = m ? acc1[j] : acc0[j];
        if (HAS_BIAS) v += bv;                          // no dead +0 adds
        if (ACT) v = v * (1.0f / (1.0f + __expf(-v)));  // SiLU
        if (TO_LDS) {
          outbuf[(m * 16 + rbase + j) * OUT_EMB + o0 + col] = (_Float16)v;
        } else {
          int nrow = node0 + m * 16 + rbase + j;
          if (nrow < N) gout[(long long)nrow * OUT_CH + o0 + col] = v;
        }
      }
    }
  }
}

// ---------------------------------------------------------------------------
// Kernel 2: node stage.  Per wave: one 32-node tile kept in LDS (f16),
// lin_up -> 3x (SiLU MLP) -> lin_out, all via v_wmma_f32_16x16x32_f16.
__global__ __launch_bounds__(64) void node_mlp_kernel(
    const float* __restrict__ agg,
    const _Float16* __restrict__ WupH, const _Float16* __restrict__ WsH,
    const _Float16* __restrict__ WoutH, const float* __restrict__ bs,
    float* __restrict__ out, int N)
{
  __shared__ _Float16 smem[2 * 2 * 32 * OUT_EMB];    // 2 waves x 2 bufs x 16KB = 64KB
  int wave = threadIdx.x >> 5;
  int lane = threadIdx.x & 31;
  int tile = blockIdx.x * 2 + wave;                  // wave-uniform
  int node0 = tile * 32;
  if (node0 >= N) return;                            // whole-wave exit: EXEC stays all-1s

  _Float16* buf0 = smem + wave * (2 * 32 * OUT_EMB);
  _Float16* buf1 = buf0 + 32 * OUT_EMB;

  // Stage agg tile [32 x 128] f32 -> f16 into buf0 (row stride OUT_EMB).
  for (int t = lane; t < 32 * HIDDEN; t += 32) {
    int r = t >> 7, c = t & (HIDDEN - 1);
    int nrow = node0 + r;
    buf0[r * OUT_EMB + c] =
        (nrow < N) ? (_Float16)agg[(long long)nrow * HIDDEN + c] : (_Float16)0.0f;
  }

  // lin_up: [32x128] @ [128->256], no bias, no act
  mlp_layer<HIDDEN,  OUT_EMB, false, false, true >(buf0, buf1, WupH, nullptr, nullptr, node0, N, lane);
  // 3 MLP layers: [32x256] @ [256->256], +bias, SiLU
  mlp_layer<OUT_EMB, OUT_EMB, true,  true,  true >(buf1, buf0, WsH + 0 * OUT_EMB * OUT_EMB, bs + 0 * OUT_EMB, nullptr, node0, N, lane);
  mlp_layer<OUT_EMB, OUT_EMB, true,  true,  true >(buf0, buf1, WsH + 1 * OUT_EMB * OUT_EMB, bs + 1 * OUT_EMB, nullptr, node0, N, lane);
  mlp_layer<OUT_EMB, OUT_EMB, true,  true,  true >(buf1, buf0, WsH + 2 * OUT_EMB * OUT_EMB, bs + 2 * OUT_EMB, nullptr, node0, N, lane);
  // lin_out: [32x256] @ [256->128], f32 straight to global
  mlp_layer<OUT_EMB, OUT_CH,  false, false, false>(buf0, nullptr, WoutH, nullptr, out, node0, N, lane);
}

// ---------------------------------------------------------------------------
extern "C" void kernel_launch(void* const* d_in, const int* in_sizes, int n_in,
                              void* d_out, int out_size, void* d_ws, size_t ws_size,
                              hipStream_t stream) {
  const float*     x    = (const float*)d_in[0];
  const float*     rbf  = (const float*)d_in[1];
  const long long* idx  = (const long long*)d_in[2];   // int64 edge indices
  const float*     Wr   = (const float*)d_in[4];       // [128,6]
  const float*     Wup  = (const float*)d_in[5];       // [256,128]
  const float*     Ws   = (const float*)d_in[6];       // [3,256,256]
  const float*     bsP  = (const float*)d_in[7];       // [3,256]
  const float*     Wout = (const float*)d_in[8];       // [128,256]
  float*           out  = (float*)d_out;

  const int E = in_sizes[0] / HIDDEN;
  const int N = out_size / OUT_CH;

  // Workspace layout
  char* ws = (char*)d_ws;
  float*    agg   = (float*)ws;                               // N*128 f32
  size_t    aggB  = (size_t)N * HIDDEN * sizeof(float);
  _Float16* WupH  = (_Float16*)(ws + aggB);                   // 256*128
  _Float16* WsH   = WupH + OUT_EMB * HIDDEN;                  // 3*256*256
  _Float16* WoutH = WsH + NUM_LAYERS * OUT_EMB * OUT_EMB;     // 128*256

  // 0) zero agg + convert weights
  prep_kernel<<<512, 256, 0, stream>>>(Wup, Ws, Wout, WupH, WsH, WoutH,
                                       agg, N * HIDDEN);

  // 1) edge fused GEMV + scatter-add (32 threads/edge, float4 x loads)
  long long totalThreads = (long long)E * 32;
  int edgeBlocks = (int)((totalThreads + 255) / 256);
  edge_scatter_kernel<<<edgeBlocks, 256, 0, stream>>>(x, rbf, idx, Wr, agg, E);

  // 2) node-side WMMA MLP: one 32-node tile per wave, 2 waves/block
  int nTiles = (N + 31) / 32;
  int mlpBlocks = (nTiles + 1) / 2;
  node_mlp_kernel<<<mlpBlocks, 64, 0, stream>>>(agg, WupH, WsH, WoutH, bsP, out, N);
}
